// PSA_47132971106545
// MI455X (gfx1250) — matrix operand order repaired and checked by reading, not compiled, not run on hardware
//
#include <hip/hip_runtime.h>
#include <hip/hip_bf16.h>

typedef __attribute__((ext_vector_type(16))) __bf16 v16bf;
typedef __attribute__((ext_vector_type(8)))  __bf16 v8bf;
typedef __attribute__((ext_vector_type(8)))  float  v8f;

#define B_ 4
#define C_ 256
#define N_ 4096
#define R_ 32

static __device__ __forceinline__ v16bf cat8(v8bf lo, v8bf hi) {
    return __builtin_shufflevector(lo, hi, 0,1,2,3,4,5,6,7,8,9,10,11,12,13,14,15);
}

// ---------------------------------------------------------------------------
// Kernel 1: fused depthwise 3x3 for dilations {1,3,5}. One block per (b,c)
// plane; x plane staged in LDS, read once. Output bf16 ydw[3][B][C][N].
// ---------------------------------------------------------------------------
__global__ __launch_bounds__(256)
void psa_dw(const float* __restrict__ x,
            const float* __restrict__ w1, const float* __restrict__ b1,
            const float* __restrict__ w2, const float* __restrict__ b2,
            const float* __restrict__ w3, const float* __restrict__ b3,
            __bf16* __restrict__ ydw)
{
    __shared__ float plane[64 * 64];
    const int bc = blockIdx.x;            // b*C + c
    const int c  = bc & (C_ - 1);
    const float* src = x + (size_t)bc * N_;
    for (int i = threadIdx.x; i < N_; i += 256) plane[i] = src[i];
    __syncthreads();

    const int dil[3] = {1, 3, 5};
    for (int i = threadIdx.x; i < N_; i += 256) {
        const int py = i >> 6, px = i & 63;
        #pragma unroll
        for (int k = 0; k < 3; ++k) {
            const float* wt = (k == 0 ? w1 : (k == 1 ? w2 : w3)) + c * 9;
            const float  bs = (k == 0 ? b1 : (k == 1 ? b2 : b3))[c];
            const int d = dil[k];
            float acc = bs;
            #pragma unroll
            for (int dy = 0; dy < 3; ++dy) {
                #pragma unroll
                for (int dx = 0; dx < 3; ++dx) {
                    const int yy = py + (dy - 1) * d;
                    const int xx = px + (dx - 1) * d;
                    if (yy >= 0 && yy < 64 && xx >= 0 && xx < 64)
                        acc += wt[dy * 3 + dx] * plane[yy * 64 + xx];
                }
            }
            ydw[(size_t)k * B_ * C_ * N_ + (size_t)bc * N_ + i] = (__bf16)acc;
        }
    }
}

// ---------------------------------------------------------------------------
// Kernel 2: pointwise 1x1, C=256 -> R=32.  Thread per (k,b,p); weights are
// wave-uniform (scalar loads).  Fp2/Fp3 written position-major [b][p][32]
// (WMMA A/B fragments become b128 loads); Fp1 written channel-major [b][r][p]
// (PV-matmul A fragment becomes b128 loads).
// ---------------------------------------------------------------------------
__global__ __launch_bounds__(256)
void psa_pw(const __bf16* __restrict__ ydw,
            const float* __restrict__ w1, const float* __restrict__ b1,
            const float* __restrict__ w2, const float* __restrict__ b2,
            const float* __restrict__ w3, const float* __restrict__ b3,
            __bf16* __restrict__ fp1R, __bf16* __restrict__ fp2T,
            __bf16* __restrict__ fp3T)
{
    const int k  = blockIdx.y;
    const int gp = blockIdx.x * 256 + threadIdx.x;   // [0, B*N)
    const int b  = gp >> 12;
    const int p  = gp & (N_ - 1);

    const float* W  = (k == 0 ? w1 : (k == 1 ? w2 : w3));
    const float* Bs = (k == 0 ? b1 : (k == 1 ? b2 : b3));

    float acc[R_];
    #pragma unroll
    for (int r = 0; r < R_; ++r) acc[r] = Bs[r];

    const __bf16* src = ydw + (size_t)k * B_ * C_ * N_ + (size_t)b * C_ * N_ + p;
    for (int c = 0; c < C_; ++c) {
        const float v = (float)src[(size_t)c * N_];
        #pragma unroll
        for (int r = 0; r < R_; ++r) acc[r] += W[r * C_ + c] * v;
    }

    if (k == 0) {
        #pragma unroll
        for (int r = 0; r < R_; ++r)
            fp1R[((size_t)b * R_ + r) * N_ + p] = (__bf16)acc[r];
    } else {
        __bf16* dst = (k == 1 ? fp2T : fp3T) + ((size_t)b * N_ + p) * R_;
        #pragma unroll
        for (int r = 0; r < R_; ++r) dst[r] = (__bf16)acc[r];
    }
}

// ---------------------------------------------------------------------------
// Kernel 3: flash attention.  4 waves/block, each wave owns one 16-row i-tile.
// j processed in 64-wide chunks: 4 score WMMAs -> online softmax (reductions
// amortized over 64 columns) -> P staged to wave-private LDS -> 4 PV WMMAs
// (O kept transposed, 32c x 16i, so row rescale / 1/L are per-lane scalars).
// All LDS traffic is wave-local: hardware keeps DS in order per wave, so only
// a compiler scheduling fence (wave_barrier) is needed -- no s_barrier.
// ---------------------------------------------------------------------------
__global__ __launch_bounds__(128)
void psa_attn(const __bf16* __restrict__ fp1R,
              const __bf16* __restrict__ fp2T,
              const __bf16* __restrict__ fp3T,
              float* __restrict__ fpmap)          // [b][i][32] f32
{
    __shared__ __attribute__((aligned(16))) __bf16 ldsP[4][16][72]; // P tile, padded
    __shared__ float ldsS[4][16];                                   // row rescale
    __shared__ float ldsL[4][16];                                   // row sum

    const int lane = threadIdx.x & 31;
    const int wv   = threadIdx.x >> 5;
    const int tile = blockIdx.x * 4 + wv;
    const int b    = tile >> 8;
    const int i0   = (tile & 255) << 4;
    const int m    = lane & 15;
    const int h    = lane >> 4;

    // A fragment: rows i0..i0+15 of Fp2^T (M=i, K=r).  Contiguous K layout:
    // lane m holds K = 16h..16h+15.
    const __bf16* arow = fp2T + ((size_t)b * N_ + i0 + m) * R_;
    const v16bf a2 = cat8(*(const v8bf*)(arow + 16 * h),
                          *(const v8bf*)(arow + 16 * h + 8));

    const v8f zc = {0.f,0.f,0.f,0.f,0.f,0.f,0.f,0.f};
    v8f o0 = zc, o1 = zc;
    float Mx[8], L[8];
    #pragma unroll
    for (int v = 0; v < 8; ++v) { Mx[v] = -3.0e38f; L[v] = 0.f; }

    for (int j0 = 0; j0 < N_; j0 += 64) {
        // ---- scores: S(16x64) = Fp2^T x Fp3 : 4 WMMAs, shared A ----------
        const __bf16* bbase = fp3T + ((size_t)b * N_ + j0 + m) * R_ + 16 * h;
        v8f s[4];
        #pragma unroll
        for (int t = 0; t < 4; ++t) {
            const __bf16* bp = bbase + (size_t)t * 16 * R_;
            const v16bf bm = cat8(*(const v8bf*)(bp), *(const v8bf*)(bp + 8));
            s[t] = __builtin_amdgcn_wmma_f32_16x16x32_bf16(false, a2, false, bm,
                                                           (short)0, zc, false, false);
        }
        if (j0 + 64 < N_) {
            __builtin_prefetch(bbase + (size_t)64 * R_, 0, 1);        // next Fp3
            __builtin_prefetch(fp1R + ((size_t)b * R_ + m) * N_ + j0 + 64, 0, 1);
        }

        // ---- online softmax; D element (v,lane): row i = v+8h, col = m ----
        float sc8[8];
        #pragma unroll
        for (int v = 0; v < 8; ++v) {
            float cm = fmaxf(fmaxf(s[0][v], s[1][v]), fmaxf(s[2][v], s[3][v]));
            cm = fmaxf(cm, __shfl_xor(cm, 1, 32));
            cm = fmaxf(cm, __shfl_xor(cm, 2, 32));
            cm = fmaxf(cm, __shfl_xor(cm, 4, 32));
            cm = fmaxf(cm, __shfl_xor(cm, 8, 32));
            const float Mn = fmaxf(Mx[v], cm);
            sc8[v] = __expf(Mx[v] - Mn);
            Mx[v]  = Mn;
            float e0 = __expf(s[0][v] - Mn);
            float e1 = __expf(s[1][v] - Mn);
            float e2 = __expf(s[2][v] - Mn);
            float e3 = __expf(s[3][v] - Mn);
            float rs = (e0 + e1) + (e2 + e3);
            rs += __shfl_xor(rs, 1, 32);
            rs += __shfl_xor(rs, 2, 32);
            rs += __shfl_xor(rs, 4, 32);
            rs += __shfl_xor(rs, 8, 32);
            L[v] = L[v] * sc8[v] + rs;
            ldsP[wv][v + 8 * h][m]      = (__bf16)e0;
            ldsP[wv][v + 8 * h][16 + m] = (__bf16)e1;
            ldsP[wv][v + 8 * h][32 + m] = (__bf16)e2;
            ldsP[wv][v + 8 * h][48 + m] = (__bf16)e3;
        }
        if (m == 0) {
            #pragma unroll
            for (int v = 0; v < 8; ++v) ldsS[wv][v + 8 * h] = sc8[v];
        }
        __builtin_amdgcn_wave_barrier();   // wave-local LDS: DS is in-order

        // ---- rescale running output: O' element (v,lane): c = v+8h, i = m -
        const float rsc = ldsS[wv][m];
        #pragma unroll
        for (int v = 0; v < 8; ++v) { o0[v] *= rsc; o1[v] *= rsc; }

        // ---- PV: O'(32c x 16i) += Fp1(32c x 64j) x P^T(64j x 16i) ---------
        const __bf16* a1p = fp1R + ((size_t)b * R_ + m) * N_ + j0;  // c = m
        const __bf16* a1q = a1p + (size_t)16 * N_;                  // c = 16+m
        const v16bf a1a0 = cat8(*(const v8bf*)(a1p + 8 * h),
                                *(const v8bf*)(a1p + 16 + 8 * h));
        const v16bf a1a1 = cat8(*(const v8bf*)(a1p + 32 + 8 * h),
                                *(const v8bf*)(a1p + 48 + 8 * h));
        const v16bf a1b0 = cat8(*(const v8bf*)(a1q + 8 * h),
                                *(const v8bf*)(a1q + 16 + 8 * h));
        const v16bf a1b1 = cat8(*(const v8bf*)(a1q + 32 + 8 * h),
                                *(const v8bf*)(a1q + 48 + 8 * h));
        // B fragments: P^T; lane holds column i=m, K = 8h..8h+7, 16+8h..23+8h (+32)
        const v16bf bP0 = cat8(*(const v8bf*)(&ldsP[wv][m][8 * h]),
                               *(const v8bf*)(&ldsP[wv][m][16 + 8 * h]));
        const v16bf bP1 = cat8(*(const v8bf*)(&ldsP[wv][m][32 + 8 * h]),
                               *(const v8bf*)(&ldsP[wv][m][48 + 8 * h]));

        o0 = __builtin_amdgcn_wmma_f32_16x16x32_bf16(false, a1a0, false, bP0,
                                                     (short)0, o0, false, false);
        o1 = __builtin_amdgcn_wmma_f32_16x16x32_bf16(false, a1b0, false, bP0,
                                                     (short)0, o1, false, false);
        o0 = __builtin_amdgcn_wmma_f32_16x16x32_bf16(false, a1a1, false, bP1,
                                                     (short)0, o0, false, false);
        o1 = __builtin_amdgcn_wmma_f32_16x16x32_bf16(false, a1b1, false, bP1,
                                                     (short)0, o1, false, false);
        __builtin_amdgcn_wave_barrier();   // keep next chunk's P stores after reads
    }

    if (m == 0) {
        #pragma unroll
        for (int v = 0; v < 8; ++v) ldsL[wv][v + 8 * h] = L[v];
    }
    __builtin_amdgcn_wave_barrier();
    const float invL = 1.0f / ldsL[wv][m];

    // store: fpmap[b][i0+m][c] = O'[c][m] / L
    float* orow = fpmap + ((size_t)b * N_ + i0 + m) * R_;
    #pragma unroll
    for (int v = 0; v < 8; ++v) {
        orow[v + 8 * h]      = o0[v] * invL;
        orow[16 + v + 8 * h] = o1[v] * invL;
    }
}

// ---------------------------------------------------------------------------
// Kernel 4: 1x1 projection 32->256 + bias + residual.  Block per (b,p).
// ---------------------------------------------------------------------------
__global__ __launch_bounds__(256)
void psa_out(const float* __restrict__ x,
             const float* __restrict__ co_w, const float* __restrict__ co_b,
             const float* __restrict__ fpmap, float* __restrict__ out)
{
    __shared__ float fr[R_];
    const int bp = blockIdx.x;                 // b*N + p
    if (threadIdx.x < R_) fr[threadIdx.x] = fpmap[(size_t)bp * R_ + threadIdx.x];
    __syncthreads();

    const int ch = threadIdx.x;
    const int b  = bp >> 12;
    const size_t xi = ((size_t)b * C_ + ch) * N_ + (bp & (N_ - 1));
    float acc = co_b[ch];
    const float* wr = co_w + ch * R_;
    #pragma unroll
    for (int r = 0; r < R_; ++r) acc += wr[r] * fr[r];
    out[xi] = x[xi] + acc;
}

// ---------------------------------------------------------------------------
extern "C" void kernel_launch(void* const* d_in, const int* in_sizes, int n_in,
                              void* d_out, int out_size, void* d_ws, size_t ws_size,
                              hipStream_t stream)
{
    const float* x     = (const float*)d_in[0];
    const float* c1dww = (const float*)d_in[1];
    const float* c1dwb = (const float*)d_in[2];
    const float* c1pww = (const float*)d_in[3];
    const float* c1pwb = (const float*)d_in[4];
    const float* c2dww = (const float*)d_in[5];
    const float* c2dwb = (const float*)d_in[6];
    const float* c2pww = (const float*)d_in[7];
    const float* c2pwb = (const float*)d_in[8];
    const float* c3dww = (const float*)d_in[9];
    const float* c3dwb = (const float*)d_in[10];
    const float* c3pww = (const float*)d_in[11];
    const float* c3pwb = (const float*)d_in[12];
    const float* co_w  = (const float*)d_in[13];
    const float* co_b  = (const float*)d_in[14];
    float* out = (float*)d_out;

    char* ws = (char*)d_ws;
    size_t off = 0;
    __bf16* ydw  = (__bf16*)(ws + off); off += (size_t)3 * B_ * C_ * N_ * 2;
    __bf16* fp1R = (__bf16*)(ws + off); off += (size_t)B_ * N_ * R_ * 2;
    __bf16* fp2T = (__bf16*)(ws + off); off += (size_t)B_ * N_ * R_ * 2;
    __bf16* fp3T = (__bf16*)(ws + off); off += (size_t)B_ * N_ * R_ * 2;
    float*  fpmp = (float*) (ws + off); off += (size_t)B_ * N_ * R_ * 4;
    (void)ws_size; (void)in_sizes; (void)n_in; (void)out_size;

    psa_dw<<<B_ * C_, 256, 0, stream>>>(x, c1dww, c1dwb, c2dww, c2dwb,
                                        c3dww, c3dwb, ydw);
    psa_pw<<<dim3(B_ * N_ / 256, 3), 256, 0, stream>>>(ydw,
                                        c1pww, c1pwb, c2pww, c2pwb,
                                        c3pww, c3pwb, fp1R, fp2T, fp3T);
    psa_attn<<<B_ * 256 / 4, 128, 0, stream>>>(fp1R, fp2T, fp3T, fpmp);
    psa_out<<<B_ * N_, 256, 0, stream>>>(x, co_w, co_b, fpmp, out);
}
